// B_TARNN_LSTM_2001454760294
// MI455X (gfx1250) — matrix operand order repaired and checked by reading, not compiled, not run on hardware
//
#include <hip/hip_runtime.h>
#include <hip/hip_bf16.h>
#include <stdint.h>

// ---------------------------------------------------------------------------
// TARNN-LSTM for MI455X (gfx1250, wave32, WMMA).
//   T=512, B=128, I=256, H=256, 2H=512, 8H=2048
// Phase 0: convert inputs + weights f32 -> bf16 (RNE).
// Phase 1: big parallel WMMA GEMM: pre[T,B,8H], bx[T,B,2H].
// Phase 2: persistent recurrent kernel, 256 blocks x 1 wave, grid barrier
//          between timesteps, weights pinned in LDS (80KB/block).
// ---------------------------------------------------------------------------

typedef unsigned short u16;
typedef __attribute__((ext_vector_type(8)))  float  v8f;
typedef __attribute__((ext_vector_type(8)))  __bf16 v8bf;
typedef __attribute__((ext_vector_type(16))) __bf16 v16bf;

#define TT   512
#define BB   128
#define II   256
#define HH   256
#define H2   512   // 2H
#define G8   2048  // 8H
#define OUT0 (TT * BB * HH)            // offset of hf in d_out
#define OUT1 (OUT0 + BB * HH)          // offset of cf in d_out
#define NBLK 256                       // persistent-kernel grid size

__device__ __forceinline__ u16 f2bf(float f) {
    unsigned u = __float_as_uint(f);
    unsigned r = ((u >> 16) & 1u) + 0x7FFFu;   // round-to-nearest-even
    return (u16)((u + r) >> 16);
}

__device__ __forceinline__ float sigm(float x)  { return 1.0f / (1.0f + __expf(-x)); }
__device__ __forceinline__ float tanh_f(float x){ return 1.0f - 2.0f / (__expf(2.0f * x) + 1.0f); }

__device__ __forceinline__ v8f wmma_bf16(v16bf a, v16bf b, v8f c) {
    // D = A(16x32 bf16) * B(32x16 bf16) + C(16x16 f32)
    return __builtin_amdgcn_wmma_f32_16x16x32_bf16(
        /*neg_a=*/false, a, /*neg_b=*/false, b,
        /*c_mod=*/(short)0, c, /*reuse_a=*/false, /*reuse_b=*/false);
}

// A tile (16x32 bf16, rows = M): per-lane one 32B chunk.
// lane<16 : row=lane,    K {0..15}
// lane>=16: row=lane-16, K {16..31}
__device__ __forceinline__ v16bf load_a_tile(const u16* base, int ld, int kb,
                                             int arow, int ak1) {
    const u16* p = base + (size_t)arow * ld + kb + ak1;
    return *(const v16bf*)p;
}

// B tile (32x16 bf16, rows = K, cols = N) built from rows of W ([N,K] row-major):
// lane<16 : col=lane,    K {0..15}   -> 32 contiguous bytes of W row
// lane>=16: col=lane-16, K {16..31}
__device__ __forceinline__ v16bf load_b_tile(const u16* w, int ld, int nb, int kb,
                                             int brow, int bko) {
    const u16* p = w + (size_t)(nb + brow) * ld + kb + bko;
    return *(const v16bf*)p;
}

// ------------------------------- conversions -------------------------------
__global__ void cvt_f32_bf16(const float* __restrict__ s, u16* __restrict__ d, int n) {
    for (int i = blockIdx.x * blockDim.x + threadIdx.x; i < n; i += gridDim.x * blockDim.x)
        d[i] = f2bf(s[i]);
}

__global__ void init_state(const float* __restrict__ h0, const float* __restrict__ c0,
                           float* __restrict__ h32, u16* __restrict__ hbf,
                           unsigned* __restrict__ bar) {
    int idx = blockIdx.x * blockDim.x + threadIdx.x;
    if (idx == 0) { bar[0] = 0u; bar[1] = 0u; }
    if (idx < BB * H2) {
        int b = idx >> 9, j = idx & 511;
        float v = (j < HH) ? h0[b * HH + j] : c0[b * HH + (j - HH)];
        h32[idx] = v;
        hbf[idx] = f2bf(v);
    }
}

// ------------------------------ phase 1 GEMM -------------------------------
// grid = (160, 4096): ct<128 -> pre column tiles, ct>=128 -> bx column tiles.
// One wave per 16x16 output tile; direct global loads (X and W stay L2-hot).
__global__ __launch_bounds__(32)
void pregemm(const u16* __restrict__ Xb,
             const u16* __restrict__ Wih, const u16* __restrict__ Wtp,
             const u16* __restrict__ Wbx,
             const float* __restrict__ bias_ih, const float* __restrict__ bias_hh,
             float* __restrict__ preO, float* __restrict__ bxO) {
    int ct = blockIdx.x, rt = blockIdx.y;
    int rbase = rt * 16;
    int lane = threadIdx.x;
    int arow = lane & 15;
    int ak1  = (lane & 16) ? 16 : 0;
    int brow = lane & 15;
    int bko  = (lane & 16) ? 16 : 0;
    int n    = lane & 15;

    v8f acc = {};
    const u16* Xc = Xb + (size_t)rbase * II;
    if (ct < 128) {
        int nb = ct * 16;
        int t  = rbase >> 7;                          // 128 rows per timestep
        const u16* Xo = Xb + (size_t)(t ? (rbase - BB) : rbase) * II;  // old_inputs
#pragma unroll
        for (int kb = 0; kb < II; kb += 32) {
            v16bf a = load_a_tile(Xc, II, kb, arow, ak1);
            v16bf b = load_b_tile(Wih, II, nb, kb, brow, bko);
            acc = wmma_bf16(a, b, acc);
        }
#pragma unroll
        for (int kb = 0; kb < II; kb += 32) {
            v16bf a = load_a_tile(Xo, II, kb, arow, ak1);
            v16bf b = load_b_tile(Wtp, II, nb, kb, brow, bko);
            acc = wmma_bf16(a, b, acc);
        }
        int col  = nb + n;
        float bi = bias_ih[col] + bias_hh[col];
#pragma unroll
        for (int i = 0; i < 8; i++) {
            int m = i + ((lane & 16) ? 8 : 0);
            preO[(size_t)(rbase + m) * G8 + col] = acc[i] + bi;
        }
    } else {
        int nb = (ct - 128) * 16;
#pragma unroll
        for (int kb = 0; kb < II; kb += 32) {
            v16bf a = load_a_tile(Xc, II, kb, arow, ak1);
            v16bf b = load_b_tile(Wbx, II, nb, kb, brow, bko);
            acc = wmma_bf16(a, b, acc);
        }
#pragma unroll
        for (int i = 0; i < 8; i++) {
            int m = i + ((lane & 16) ? 8 : 0);
            bxO[(size_t)(rbase + m) * H2 + nb + n] = acc[i];
        }
    }
}

// --------------------------- device-wide barrier ---------------------------
__device__ __forceinline__ void grid_sync(unsigned* bar, unsigned target) {
    __threadfence();
    if (threadIdx.x == 0) {
        unsigned prev = atomicAdd(&bar[0], 1u);
        if (prev == NBLK - 1u) {
            __hip_atomic_store(&bar[0], 0u, __ATOMIC_RELAXED, __HIP_MEMORY_SCOPE_AGENT);
            __threadfence();
            atomicAdd(&bar[1], 1u);
        } else {
            while (__hip_atomic_load(&bar[1], __ATOMIC_ACQUIRE,
                                     __HIP_MEMORY_SCOPE_AGENT) < target) {
                __builtin_amdgcn_s_sleep(8);
            }
        }
    }
    __syncthreads();
    __threadfence();
}

// ---------------------------- phase 2 recurrent ----------------------------
// 256 blocks x 32 threads, persistent over all T=512 steps.
// Block (mt,nt) owns output tile rows [mt*16,+16) of B, cols [nt*16,+16) of 2H.
// Per step: 5 GEMMs of [16,512]x[512,16]:
//   g=0..3 -> gates (W_hh rows g*512 + cols), g=4 -> W_bh. 80 WMMAs/step/block.
// Weights staged once in 80KB LDS; h ping-pongs in global (f32 + bf16 copies).
__global__ __launch_bounds__(32)
void recurrent(const u16* __restrict__ Whh, const u16* __restrict__ Wbh,
               const float* __restrict__ preA, const float* __restrict__ bxA,
               float* __restrict__ h32, u16* __restrict__ hbf,
               float* __restrict__ out, unsigned* __restrict__ bar) {
    extern __shared__ u16 lds[];                 // 5 * 16 rows * 512 bf16 = 80KB
    int bid = blockIdx.x;
    int nt = bid & 31, mt = bid >> 5;
    int n_base = nt * 16, m_base = mt * 16;
    int lane = threadIdx.x;

    // Stage this block's W_hh gate slices + W_bh slice into LDS (uint4 copies).
    {
        uint4* ldq = (uint4*)lds;
        for (int idx = lane; idx < 5 * 16 * 64; idx += 32) {   // 64 uint4 per row
            int row = idx >> 6, c = idx & 63;
            int g = row >> 4, r = row & 15;
            const u16* src = (g < 4)
                ? (Whh + (size_t)(g * H2 + n_base + r) * H2)
                : (Wbh + (size_t)(n_base + r) * H2);
            ldq[idx] = *(const uint4*)(src + c * 8);
        }
    }
    __syncthreads();

    int arow = lane & 15;
    int ak1  = (lane & 16) ? 16 : 0;
    int brow = lane & 15;
    int bko  = (lane & 16) ? 16 : 0;
    int n    = lane & 15;

    for (int t = 0; t < TT; t++) {
        int cur = t & 1, nxt = cur ^ 1;
        const u16* hb = hbf + (size_t)cur * (BB * H2);

        v8f acc0 = {}, acc1 = {}, acc2 = {}, acc3 = {}, acc4 = {};
#pragma unroll 4
        for (int kb = 0; kb < H2; kb += 32) {
            v16bf a = load_a_tile(hb + (size_t)m_base * H2, H2, kb, arow, ak1);
            const u16* lb = lds + (size_t)brow * H2 + kb + bko;
            acc0 = wmma_bf16(a, *(const v16bf*)(lb + 0 * 16 * H2), acc0);
            acc1 = wmma_bf16(a, *(const v16bf*)(lb + 1 * 16 * H2), acc1);
            acc2 = wmma_bf16(a, *(const v16bf*)(lb + 2 * 16 * H2), acc2);
            acc3 = wmma_bf16(a, *(const v16bf*)(lb + 3 * 16 * H2), acc3);
            acc4 = wmma_bf16(a, *(const v16bf*)(lb + 4 * 16 * H2), acc4);
        }

        const float* preb = preA + (size_t)t * BB * G8;
        const float* bxb  = bxA  + (size_t)t * BB * H2;
        const float* hcur = h32 + (size_t)cur * (BB * H2);
        float* hnx  = h32 + (size_t)nxt * (BB * H2);
        u16*   hbnx = hbf + (size_t)nxt * (BB * H2);

#pragma unroll
        for (int i = 0; i < 8; i++) {
            int m = i + ((lane & 16) ? 8 : 0);
            int b = m_base + m;
            int j = n_base + n;
            const float* pr = preb + (size_t)b * G8;
            float ig = sigm  (acc0[i] + pr[j]);
            float fg = sigm  (acc1[i] + pr[H2 + j]);
            float cg = tanh_f(acc2[i] + pr[2 * H2 + j]);
            float og = sigm  (acc3[i] + pr[3 * H2 + j]);
            float hold = hcur[b * H2 + j];
            float hn = fg * hold + ig * (cg + bxb[b * H2 + j] + acc4[i]);
            hn = og * tanh_f(hn);
            hnx[b * H2 + j]  = hn;
            hbnx[b * H2 + j] = f2bf(hn);
            if (j < HH)                                         // uniform per block
                out[((size_t)t * BB + b) * HH + j] = hn;
            if (t == TT - 1) {
                if (j < HH) out[OUT0 + b * HH + j] = hn;        // hf
                else        out[OUT1 + b * HH + (j - HH)] = hn; // cf
            }
        }

        if (t + 1 < TT)  // warm next step's pre tile (global_prefetch_b8)
            __builtin_prefetch(preA + ((size_t)(t + 1) * BB + m_base) * G8 + n_base, 0, 1);

        grid_sync(bar, (unsigned)(t + 1));
    }
}

// -------------------------------- launcher ---------------------------------
extern "C" void kernel_launch(void* const* d_in, const int* in_sizes, int n_in,
                              void* d_out, int out_size, void* d_ws, size_t ws_size,
                              hipStream_t stream) {
    const float* inputs  = (const float*)d_in[0];
    const float* h0      = (const float*)d_in[1];
    const float* c0      = (const float*)d_in[2];
    const float* w_ih    = (const float*)d_in[3];
    const float* w_hh    = (const float*)d_in[4];
    const float* bias_ih = (const float*)d_in[5];
    const float* bias_hh = (const float*)d_in[6];
    const float* w_bx    = (const float*)d_in[7];
    const float* w_bh    = (const float*)d_in[8];
    const float* w_tp    = (const float*)d_in[9];
    float* out = (float*)d_out;

    // Workspace carve (needs ~711 MB; pre[T,B,8H] f32 dominates).
    uint8_t* w = (uint8_t*)d_ws;
    size_t off = 0;
    auto carve = [&](size_t bytes) -> void* {
        void* p = w + off;
        off += (bytes + 255) & ~(size_t)255;
        return p;
    };
    u16*   Xb   = (u16*)  carve((size_t)TT * BB * II * 2);
    u16*   Wihb = (u16*)  carve((size_t)G8 * II * 2);
    u16*   Wtpb = (u16*)  carve((size_t)G8 * II * 2);
    u16*   Wbxb = (u16*)  carve((size_t)H2 * II * 2);
    u16*   Whhb = (u16*)  carve((size_t)G8 * H2 * 2);
    u16*   Wbhb = (u16*)  carve((size_t)H2 * H2 * 2);
    float* pre  = (float*)carve((size_t)TT * BB * G8 * 4);
    float* bx   = (float*)carve((size_t)TT * BB * H2 * 4);
    float* h32  = (float*)carve((size_t)2 * BB * H2 * 4);
    u16*   hbf  = (u16*)  carve((size_t)2 * BB * H2 * 2);
    unsigned* bar = (unsigned*)carve(256);
    (void)ws_size; (void)in_sizes; (void)n_in; (void)out_size;

    // Phase 0: bf16 conversions + state/barrier init.
    cvt_f32_bf16<<<4096, 256, 0, stream>>>(inputs, Xb, TT * BB * II);
    cvt_f32_bf16<<<1024, 256, 0, stream>>>(w_ih, Wihb, G8 * II);
    cvt_f32_bf16<<<1024, 256, 0, stream>>>(w_tp, Wtpb, G8 * II);
    cvt_f32_bf16<<<512,  256, 0, stream>>>(w_bx, Wbxb, H2 * II);
    cvt_f32_bf16<<<1024, 256, 0, stream>>>(w_hh, Whhb, G8 * H2);
    cvt_f32_bf16<<<1024, 256, 0, stream>>>(w_bh, Wbhb, H2 * H2);
    init_state<<<(BB * H2 + 255) / 256, 256, 0, stream>>>(h0, c0, h32, hbf, bar);

    // Phase 1: parallel input-side GEMMs (pre + bx).
    pregemm<<<dim3(160, (TT * BB) / 16), 32, 0, stream>>>(
        Xb, Wihb, Wtpb, Wbxb, bias_ih, bias_hh, pre, bx);

    // Phase 2: persistent recurrent scan, 80KB dynamic LDS per block.
    recurrent<<<NBLK, 32, 5 * 16 * H2 * sizeof(u16), stream>>>(
        Whhb, Wbhb, pre, bx, h32, hbf, out, bar);
}
